// PathwayAwareTokenizer_65652870086818
// MI455X (gfx1250) — matrix operand order repaired and checked by reading, not compiled, not run on hardware
//
#include <hip/hip_runtime.h>
#include <hip/hip_bf16.h>
#include <math.h>

typedef _Float16 f16;
typedef __attribute__((ext_vector_type(8)))  f16   v8h;
typedef __attribute__((ext_vector_type(16))) f16   v16h;
typedef __attribute__((ext_vector_type(8)))  float v8f;

// ---------------------------------------------------------------------------
// WMMA helpers (CDNA5 gfx1250, wave32)
// ---------------------------------------------------------------------------
__device__ __forceinline__ v8f wmma_f16(v16h a, v16h b, v8f c) {
  // D = A(16x32 f16) x B(32x16 f16) + C(16x16 f32)
  return __builtin_amdgcn_wmma_f32_16x16x32_f16(false, a, false, b, (short)0, c,
                                                false, false);
}

// A-matrix fragment (16x32 f16). Lane L: row = base + (L&15), half = L>>4.
// halfs[0..7]  = K[half*8 .. half*8+7]
// halfs[8..15] = K[16 + half*8 .. 16 + half*8 + 7]
__device__ __forceinline__ v16h ldfragA(const f16* rowptr, int k0, int hf) {
  v8h lo = *(const v8h*)(rowptr + k0 + hf * 8);
  v8h hi = *(const v8h*)(rowptr + k0 + 16 + hf * 8);
  v16h o;
#pragma unroll
  for (int i = 0; i < 8; ++i) { o[i] = lo[i]; o[i + 8] = hi[i]; }
  return o;
}

// B-matrix fragment (32x16 f16), built from row-major W[n][k].
// Lane L: col n = base + (L&15), half-group hf = L>>4.
// halfs[0..7] = K[hf*8 .. hf*8+7], halfs[8..15] = K[16 + hf*8 .. 16 + hf*8 + 7]
__device__ __forceinline__ v16h ldfragB(const f16* rowptr, int k0, int hf) {
  v8h lo = *(const v8h*)(rowptr + k0 + hf * 8);
  v8h hi = *(const v8h*)(rowptr + k0 + 16 + hf * 8);
  v16h o;
#pragma unroll
  for (int i = 0; i < 8; ++i) { o[i] = lo[i]; o[i + 8] = hi[i]; }
  return o;
}

// ---------------------------------------------------------------------------
// f32 -> f16 convert with zero padding: src [M,K] -> dst [Mpad,Kpad]
// ---------------------------------------------------------------------------
__global__ void k_cvt_pad(const float* __restrict__ src, f16* __restrict__ dst,
                          int M, int K, int Mpad, int Kpad) {
  long long i = (long long)blockIdx.x * blockDim.x + threadIdx.x;
  long long tot = (long long)Mpad * Kpad;
  if (i >= tot) return;
  int k = (int)(i % Kpad);
  long long m = i / Kpad;
  float v = 0.f;
  if (m < M && k < K) v = src[m * (long long)K + k];
  dst[i] = (f16)v;
}

// ---------------------------------------------------------------------------
// WMMA GEMM: C[M,Nvalid] = A(f16 [M,Kp]) x W(f16 [Npad,Kp])^T + bias
// BM=128 BN=64 BK=32, 256 threads (8 waves), wave w owns rows [w*16, w*16+16)
// ---------------------------------------------------------------------------
__global__ __launch_bounds__(256) void k_wmma_gemm(
    const f16* __restrict__ A, const f16* __restrict__ W,
    const float* __restrict__ bias, float* __restrict__ C,
    int M, int Kp, int Npad, int Nvalid, int ldc) {
  __shared__ f16 As[128][32];
  __shared__ f16 Bs[64][32];

  const int tid = threadIdx.x;
  const int wave = tid >> 5, lane = tid & 31;
  const int r = lane & 15, hf = lane >> 4;
  const int m0 = blockIdx.x * 128;
  const int n0 = blockIdx.y * 64;

  v8f acc[4] = {};

  for (int k0 = 0; k0 < Kp; k0 += 32) {
    __syncthreads();
    {  // stage A: 128x32 halfs; each thread loads 16 contiguous halfs
      int row = tid >> 1, seg = tid & 1;
      v8h a0 = {}, a1 = {};
      if (m0 + row < M) {
        const f16* src = A + (size_t)(m0 + row) * Kp + k0 + seg * 16;
        a0 = ((const v8h*)src)[0];
        a1 = ((const v8h*)src)[1];
      }
      v8h* dst = (v8h*)&As[row][seg * 16];
      dst[0] = a0; dst[1] = a1;
    }
    {  // stage B: 64x32 halfs; each thread loads 8 halfs
      int n = tid >> 2, seg = tid & 3;
      v8h v = {};
      if (n0 + n < Npad) v = *(const v8h*)(W + (size_t)(n0 + n) * Kp + k0 + seg * 8);
      *(v8h*)&Bs[n][seg * 8] = v;
    }
    __syncthreads();

    v16h af = ldfragA(&As[wave * 16 + r][0], 0, hf);
#pragma unroll
    for (int nt = 0; nt < 4; ++nt) {
      v16h bf = ldfragB(&Bs[nt * 16 + r][0], 0, hf);
      acc[nt] = wmma_f16(af, bf, acc[nt]);
    }
  }

  // epilogue: col = n0 + nt*16 + lane%16, rows = (lane/16)*8 + q
#pragma unroll
  for (int nt = 0; nt < 4; ++nt) {
    int col = n0 + nt * 16 + r;
    if (col < Nvalid) {
      float bv = bias ? bias[col] : 0.f;
      int mb = m0 + wave * 16 + hf * 8;
#pragma unroll
      for (int q = 0; q < 8; ++q)
        if (mb + q < M) C[(size_t)(mb + q) * ldc + col] = acc[nt][q] + bv;
    }
  }
}

// ---------------------------------------------------------------------------
// Row-wise LayerNorm + exact GELU, f16 output. One wave per row.
// ---------------------------------------------------------------------------
__global__ __launch_bounds__(256) void k_ln_gelu(
    const float* __restrict__ X, const float* __restrict__ g,
    const float* __restrict__ bb, f16* __restrict__ Y, int M, int N) {
  int wave = threadIdx.x >> 5, lane = threadIdx.x & 31;
  int row = blockIdx.x * 8 + wave;
  if (row >= M) return;
  const float* x = X + (size_t)row * N;
  float s = 0.f, s2 = 0.f;
  for (int c = lane; c < N; c += 32) { float v = x[c]; s += v; s2 += v * v; }
#pragma unroll
  for (int m = 16; m > 0; m >>= 1) {
    s += __shfl_xor(s, m, 32);
    s2 += __shfl_xor(s2, m, 32);
  }
  float mean = s / N;
  float var = s2 / N - mean * mean;
  float inv = rsqrtf(var + 1e-5f);
  f16* y = Y + (size_t)row * N;
  for (int c = lane; c < N; c += 32) {
    float v = (x[c] - mean) * inv * g[c] + bb[c];
    float ge = 0.5f * v * (1.f + erff(v * 0.70710678118654752f));
    y[c] = (f16)ge;
  }
}

// ---------------------------------------------------------------------------
// softmax over first 82 of a 96-wide row. One wave per row.
// ---------------------------------------------------------------------------
__global__ __launch_bounds__(256) void k_softmax82(
    const float* __restrict__ X, float* __restrict__ O, int M) {
  int wave = threadIdx.x >> 5, lane = threadIdx.x & 31;
  int row = blockIdx.x * 8 + wave;
  if (row >= M) return;
  const float* x = X + (size_t)row * 96;
  float v[3];
  float mx = -1e30f;
#pragma unroll
  for (int q = 0; q < 3; ++q) {
    int c = lane + q * 32;
    v[q] = (c < 82) ? x[c] : -1e30f;
    mx = fmaxf(mx, v[q]);
  }
#pragma unroll
  for (int m = 16; m > 0; m >>= 1) mx = fmaxf(mx, __shfl_xor(mx, m, 32));
  float s = 0.f;
#pragma unroll
  for (int q = 0; q < 3; ++q) {
    int c = lane + q * 32;
    v[q] = (c < 82) ? __expf(v[q] - mx) : 0.f;
    s += v[q];
  }
#pragma unroll
  for (int m = 16; m > 0; m >>= 1) s += __shfl_xor(s, m, 32);
  float inv = 1.f / s;
#pragma unroll
  for (int q = 0; q < 3; ++q) {
    int c = lane + q * 32;
    if (c < 82) O[(size_t)row * 82 + c] = v[q] * inv;
  }
}

// ---------------------------------------------------------------------------
// acts[b,t,d] = (gene_w[b,t] + prot_w[b,t]) * tokens[t,d]
// ---------------------------------------------------------------------------
__global__ void k_init_acts(const float* __restrict__ gw,
                            const float* __restrict__ pw,
                            const float* __restrict__ pt,
                            const float* __restrict__ mt,
                            float* __restrict__ out, int total) {
  int i = blockIdx.x * blockDim.x + threadIdx.x;
  if (i >= total) return;
  int d = i % 24;
  int t = (i / 24) % 82;
  int b = i / (24 * 82);
  float tok = (t < 50) ? pt[t * 24 + d] : mt[(t - 50) * 24 + d];
  out[i] = (gw[b * 82 + t] + pw[b * 82 + t]) * tok;
}

// ---------------------------------------------------------------------------
// One post-norm transformer encoder layer, one block (256 thr) per batch elem.
// x: [82,24] in/out on d_out, in place. WMMA for QKV / FFN1 / FFN2.
// ---------------------------------------------------------------------------
__global__ __launch_bounds__(256) void k_encoder(
    float* __restrict__ acts,                // [B,82,24]
    const f16* __restrict__ wqkvh,           // [80,32] f16 padded
    const float* __restrict__ bqkv,          // [72]
    const float* __restrict__ wo,            // [24,24]
    const float* __restrict__ bo,            // [24]
    const float* __restrict__ lg1, const float* __restrict__ lb1,
    const f16* __restrict__ fw1h,            // [96,32]
    const float* __restrict__ fb1,           // [96]
    const f16* __restrict__ fw2h,            // [32,96]
    const float* __restrict__ fb2,           // [24]
    const float* __restrict__ lg2, const float* __restrict__ lb2) {
  __shared__ float sX[82][24];                       // residual (f32)
  __shared__ f16 sXh[96][32];                        // x, f16, padded
  __shared__ float sO[82][24];                       // attn out, later FFN2 out
  __shared__ __align__(16) unsigned char pool[42272];
  f16(*sQKV)[80] = (f16(*)[80])pool;                 // [96][80] f16 (15360 B)
  float(*sAtt)[82] = (float(*)[82])(pool + 15360);   // [82][82] f32 (26896 B)
  f16(*sHh)[96] = (f16(*)[96])pool;                  // [96][96] f16 (18432 B)
  float* sP = (float*)(pool + 15360);                // [82*24] proj scratch

  const int tid = threadIdx.x;
  const int wave = tid >> 5, lane = tid & 31;
  const int r = lane & 15, hf = lane >> 4;
  const int b = blockIdx.x;
  float* xg = acts + (size_t)b * 82 * 24;

  // ---- load x -> sX (f32) and sXh (f16, zero padded to 96x32) ----
  for (int i = tid; i < 96 * 32; i += 256) {
    int row = i >> 5, c = i & 31;
    float v = 0.f;
    if (row < 82 && c < 24) { v = xg[row * 24 + c]; sX[row][c] = v; }
    sXh[row][c] = (f16)v;
  }
  __syncthreads();

  // ---- QKV = x @ wqkv^T + bqkv : M=96, N=80, K=32 ; 30 WMMA tiles ----
  for (int j = wave; j < 30; j += 8) {
    int mt = j / 5, nt = j % 5;
    v16h af = ldfragA(&sXh[mt * 16 + r][0], 0, hf);
    v16h bf = ldfragB(wqkvh + (size_t)(nt * 16 + r) * 32, 0, hf);
    v8f acc = {};
    acc = wmma_f16(af, bf, acc);
    int n = nt * 16 + r;
    int mb = mt * 16 + hf * 8;
    float bv = (n < 72) ? bqkv[n] : 0.f;
#pragma unroll
    for (int q = 0; q < 8; ++q) sQKV[mb + q][n] = (f16)(acc[q] + bv);
  }
  __syncthreads();

  // ---- attention per head (DH=6 too skinny for WMMA -> VALU) ----
  const float scale = 0.40824829046386307f;  // 1/sqrt(6)
  for (int h = 0; h < 4; ++h) {
    int qo = h * 6, ko = 24 + h * 6, vo = 48 + h * 6;
    for (int p = tid; p < 82 * 82; p += 256) {
      int i = p / 82, j2 = p % 82;
      float s = 0.f;
#pragma unroll
      for (int d = 0; d < 6; ++d)
        s += (float)sQKV[i][qo + d] * (float)sQKV[j2][ko + d];
      sAtt[i][j2] = s * scale;
    }
    __syncthreads();
    for (int i = tid; i < 82; i += 256) {
      float mx = -1e30f;
      for (int j2 = 0; j2 < 82; ++j2) mx = fmaxf(mx, sAtt[i][j2]);
      float sm = 0.f;
      for (int j2 = 0; j2 < 82; ++j2) {
        float e = __expf(sAtt[i][j2] - mx);
        sAtt[i][j2] = e;
        sm += e;
      }
      float inv = 1.f / sm;
      for (int j2 = 0; j2 < 82; ++j2) sAtt[i][j2] *= inv;
    }
    __syncthreads();
    for (int p = tid; p < 82 * 6; p += 256) {
      int i = p / 6, d = p % 6;
      float s = 0.f;
      for (int j2 = 0; j2 < 82; ++j2) s += sAtt[i][j2] * (float)sQKV[j2][vo + d];
      sO[i][h * 6 + d] = s;
    }
    __syncthreads();
  }

  // ---- o-proj + residual -> sP ----
  for (int p = tid; p < 82 * 24; p += 256) {
    int i = p / 24, d = p % 24;
    float s = bo[d];
#pragma unroll
    for (int e = 0; e < 24; ++e) s += sO[i][e] * wo[d * 24 + e];
    sP[i * 24 + d] = sX[i][d] + s;
  }
  __syncthreads();
  // ---- LN1 -> sX (f32) and sXh (f16) ----
  for (int i = tid; i < 82; i += 256) {
    float m = 0.f;
    for (int d = 0; d < 24; ++d) m += sP[i * 24 + d];
    m *= (1.f / 24.f);
    float v = 0.f;
    for (int d = 0; d < 24; ++d) { float t = sP[i * 24 + d] - m; v += t * t; }
    v *= (1.f / 24.f);
    float inv = rsqrtf(v + 1e-5f);
    for (int d = 0; d < 24; ++d) {
      float y = (sP[i * 24 + d] - m) * inv * lg1[d] + lb1[d];
      sX[i][d] = y;
      sXh[i][d] = (f16)y;
    }
  }
  __syncthreads();

  // ---- FFN1: h = relu(x @ fw1^T + fb1) : M=96, N=96, K=32 -> f16 sHh ----
  for (int j = wave; j < 36; j += 8) {
    int mt = j / 6, nt = j % 6;
    v16h af = ldfragA(&sXh[mt * 16 + r][0], 0, hf);
    v16h bf = ldfragB(fw1h + (size_t)(nt * 16 + r) * 32, 0, hf);
    v8f acc = {};
    acc = wmma_f16(af, bf, acc);
    int n = nt * 16 + r;
    int mb = mt * 16 + hf * 8;
    float bv = fb1[n];
#pragma unroll
    for (int q = 0; q < 8; ++q) sHh[mb + q][n] = (f16)fmaxf(acc[q] + bv, 0.f);
  }
  __syncthreads();

  // ---- FFN2: f2 = h @ fw2^T : M=96, N=32, K=96 (3 k-steps) -> sO (f32) ----
  for (int j = wave; j < 12; j += 8) {
    int mt = j / 2, nt = j % 2;
    v8f acc = {};
#pragma unroll
    for (int ks = 0; ks < 3; ++ks) {
      v16h af = ldfragA(&sHh[mt * 16 + r][0], ks * 32, hf);
      v16h bf = ldfragB(fw2h + (size_t)(nt * 16 + r) * 96, ks * 32, hf);
      acc = wmma_f16(af, bf, acc);
    }
    int n = nt * 16 + r;
    if (n < 24) {
      int mb = mt * 16 + hf * 8;
#pragma unroll
      for (int q = 0; q < 8; ++q)
        if (mb + q < 82) sO[mb + q][n] = acc[q];
    }
  }
  __syncthreads();

  // ---- residual + fb2 + LN2 -> global (in place) ----
  for (int i = tid; i < 82; i += 256) {
    float tmp[24];
    float m = 0.f;
    for (int d = 0; d < 24; ++d) {
      float y = sX[i][d] + sO[i][d] + fb2[d];
      tmp[d] = y;
      m += y;
    }
    m *= (1.f / 24.f);
    float v = 0.f;
    for (int d = 0; d < 24; ++d) { float t = tmp[d] - m; v += t * t; }
    v *= (1.f / 24.f);
    float inv = rsqrtf(v + 1e-5f);
    for (int d = 0; d < 24; ++d)
      xg[i * 24 + d] = (tmp[d] - m) * inv * lg2[d] + lb2[d];
  }
}

// ---------------------------------------------------------------------------
// host launch
// ---------------------------------------------------------------------------
extern "C" void kernel_launch(void* const* d_in, const int* in_sizes, int n_in,
                              void* d_out, int out_size, void* d_ws,
                              size_t ws_size, hipStream_t stream) {
  (void)in_sizes; (void)n_in; (void)out_size; (void)ws_size;
  const int B = 8192, T = 82, D = 24;

  const float* gene = (const float*)d_in[0];
  const float* protein = (const float*)d_in[1];
  const float* ptok = (const float*)d_in[2];
  const float* mtok = (const float*)d_in[3];
  const float* gw1 = (const float*)d_in[4];  const float* gb1 = (const float*)d_in[5];
  const float* gg1 = (const float*)d_in[6];  const float* gbb1 = (const float*)d_in[7];
  const float* gw2 = (const float*)d_in[8];  const float* gb2 = (const float*)d_in[9];
  const float* gg2 = (const float*)d_in[10]; const float* gbb2 = (const float*)d_in[11];
  const float* gw3 = (const float*)d_in[12]; const float* gb3 = (const float*)d_in[13];
  const float* pw1 = (const float*)d_in[14]; const float* pb1 = (const float*)d_in[15];
  const float* pg1 = (const float*)d_in[16]; const float* pbb1 = (const float*)d_in[17];
  const float* pw2 = (const float*)d_in[18]; const float* pb2 = (const float*)d_in[19];
  const float* pg2 = (const float*)d_in[20]; const float* pbb2 = (const float*)d_in[21];
  const float* pw3 = (const float*)d_in[22]; const float* pb3 = (const float*)d_in[23];
  const float* wqkv = (const float*)d_in[24]; const float* bqkv = (const float*)d_in[25];
  const float* wo = (const float*)d_in[26];   const float* bo = (const float*)d_in[27];
  const float* lg1 = (const float*)d_in[28];  const float* lb1 = (const float*)d_in[29];
  const float* fw1 = (const float*)d_in[30];  const float* fb1 = (const float*)d_in[31];
  const float* fw2 = (const float*)d_in[32];  const float* fb2 = (const float*)d_in[33];
  const float* lg2 = (const float*)d_in[34];  const float* lb2 = (const float*)d_in[35];
  float* out = (float*)d_out;

  // ---- workspace bump allocator ----
  char* wsb = (char*)d_ws;
  size_t off = 0;
  auto alloc = [&](size_t bytes) -> void* {
    off = (off + 255) & ~(size_t)255;
    void* p = wsb + off;
    off += bytes;
    return p;
  };
  const int KG = 5024;  // 5000 padded to mult of 32
  f16* geneh = (f16*)alloc((size_t)B * KG * 2);
  f16* proth = (f16*)alloc((size_t)B * 256 * 2);
  f16* gw1h = (f16*)alloc((size_t)512 * KG * 2);
  f16* gw2h = (f16*)alloc((size_t)256 * 512 * 2);
  f16* gw3h = (f16*)alloc((size_t)96 * 256 * 2);
  f16* pw1h = (f16*)alloc((size_t)512 * 256 * 2);
  f16* pw2h = (f16*)alloc((size_t)256 * 512 * 2);
  f16* pw3h = (f16*)alloc((size_t)96 * 256 * 2);
  f16* wqkvh = (f16*)alloc((size_t)2 * 80 * 32 * 2);
  f16* fw1h = (f16*)alloc((size_t)2 * 96 * 32 * 2);
  f16* fw2h = (f16*)alloc((size_t)2 * 32 * 96 * 2);
  float* t512 = (float*)alloc((size_t)B * 512 * 4);
  f16* h512 = (f16*)alloc((size_t)B * 512 * 2);
  float* t256 = (float*)alloc((size_t)B * 256 * 4);
  f16* h256 = (f16*)alloc((size_t)B * 256 * 2);
  float* t96 = (float*)alloc((size_t)B * 96 * 4);
  float* gene_w = (float*)alloc((size_t)B * T * 4);
  float* prot_w = (float*)alloc((size_t)B * T * 4);

  auto cvt = [&](const float* src, f16* dst, int M, int K, int Mpad, int Kpad) {
    long long tot = (long long)Mpad * Kpad;
    int g = (int)((tot + 255) / 256);
    k_cvt_pad<<<g, 256, 0, stream>>>(src, dst, M, K, Mpad, Kpad);
  };
  auto gemm = [&](const f16* A, const f16* W, const float* bias, float* C,
                  int M, int Kp, int Npad, int Nvalid, int ldc) {
    dim3 g(M / 128, (Nvalid + 63) / 64);
    k_wmma_gemm<<<g, 256, 0, stream>>>(A, W, bias, C, M, Kp, Npad, Nvalid, ldc);
  };

  // ---- weight / activation packing to f16 ----
  cvt(gene, geneh, B, 5000, B, KG);
  cvt(protein, proth, B, 226, B, 256);
  cvt(gw1, gw1h, 512, 5000, 512, KG);
  cvt(gw2, gw2h, 256, 512, 256, 512);
  cvt(gw3, gw3h, 82, 256, 96, 256);
  cvt(pw1, pw1h, 512, 226, 512, 256);
  cvt(pw2, pw2h, 256, 512, 256, 512);
  cvt(pw3, pw3h, 82, 256, 96, 256);
  for (int l = 0; l < 2; ++l) {
    cvt(wqkv + l * 72 * 24, wqkvh + l * 80 * 32, 72, 24, 80, 32);
    cvt(fw1 + l * 96 * 24, fw1h + l * 96 * 32, 96, 24, 96, 32);
    cvt(fw2 + l * 24 * 96, fw2h + l * 32 * 96, 24, 96, 32, 96);
  }

  const int rowsBlocks = B / 8;  // ln/softmax: one wave per row

  // ---- gene adapter: 5000 -> 512 -> 256 -> 82, softmax ----
  gemm(geneh, gw1h, gb1, t512, B, KG, 512, 512, 512);
  k_ln_gelu<<<rowsBlocks, 256, 0, stream>>>(t512, gg1, gbb1, h512, B, 512);
  gemm(h512, gw2h, gb2, t256, B, 512, 256, 256, 256);
  k_ln_gelu<<<rowsBlocks, 256, 0, stream>>>(t256, gg2, gbb2, h256, B, 256);
  gemm(h256, gw3h, gb3, t96, B, 256, 96, 82, 96);
  k_softmax82<<<rowsBlocks, 256, 0, stream>>>(t96, gene_w, B);

  // ---- protein adapter (reuses scratch; stream-serialized) ----
  gemm(proth, pw1h, pb1, t512, B, 256, 512, 512, 512);
  k_ln_gelu<<<rowsBlocks, 256, 0, stream>>>(t512, pg1, pbb1, h512, B, 512);
  gemm(h512, pw2h, pb2, t256, B, 512, 256, 256, 256);
  k_ln_gelu<<<rowsBlocks, 256, 0, stream>>>(t256, pg2, pbb2, h256, B, 256);
  gemm(h256, pw3h, pb3, t96, B, 256, 96, 82, 96);
  k_softmax82<<<rowsBlocks, 256, 0, stream>>>(t96, prot_w, B);

  // ---- token activations straight into d_out ----
  int total = B * T * D;
  k_init_acts<<<(total + 255) / 256, 256, 0, stream>>>(gene_w, prot_w, ptok,
                                                       mtok, out, total);

  // ---- two encoder layers, in place on d_out ----
  for (int l = 0; l < 2; ++l) {
    k_encoder<<<B, 256, 0, stream>>>(
        out, wqkvh + l * 80 * 32, bqkv + l * 72, wo + l * 24 * 24, bo + l * 24,
        lg1 + l * 24, lb1 + l * 24, fw1h + l * 96 * 32, fb1 + l * 96,
        fw2h + l * 32 * 96, fb2 + l * 24, lg2 + l * 24, lb2 + l * 24);
  }
}
